// KernelUpdator_28621662061052
// MI455X (gfx1250) — matrix operand order repaired and checked by reading, not compiled, not run on hardware
//
#include <hip/hip_runtime.h>
#include <hip/hip_bf16.h>

#define C_DIM 256
#define EPSV  1e-5f

typedef __bf16 bf16_t;
typedef __attribute__((ext_vector_type(16))) __bf16 v16bf;
typedef __attribute__((ext_vector_type(8)))  float  v8f;

enum { MODE_PLAIN = 0, MODE_LN = 1, MODE_GATE = 2 };

// ---------------------------------------------------------------------------
// f32 -> bf16 convert (weights only; activations converted on the fly)
// ---------------------------------------------------------------------------
__global__ void cvt_f32_bf16(const float* __restrict__ in, bf16_t* __restrict__ out, int n) {
    int i = blockIdx.x * blockDim.x + threadIdx.x;
    if (i < n) out[i] = (bf16_t)in[i];
}

// ---------------------------------------------------------------------------
// Shared helpers
// ---------------------------------------------------------------------------
// 8-step K loop over C=256 with A staged fragment-ordered in LDS.
__device__ __forceinline__ void wave_gemm_256(const bf16_t* ldsA, const bf16_t* __restrict__ W,
                                              int colbase, int lane, v8f& c0, v8f& c1) {
    const int m_of = lane & 15;
    const int h_of = lane >> 4;
    #pragma unroll
    for (int ks = 0; ks < 8; ++ks) {
        v16bf a = *(const v16bf*)(ldsA + ks * 512 + lane * 16);
        const bf16_t* bp = W + (size_t)(colbase + m_of) * C_DIM + ks * 32 + h_of * 16;
        v16bf b0 = *(const v16bf*)bp;
        v16bf b1 = *(const v16bf*)(bp + 16 * C_DIM);
        c0 = __builtin_amdgcn_wmma_f32_16x16x32_bf16(false, a, false, b0, (short)0, c0, false, false);
        c1 = __builtin_amdgcn_wmma_f32_16x16x32_bf16(false, a, false, b1, (short)0, c1, false, false);
    }
}

// Scatter f32 accumulators (+bias) into a 16x256 LDS tile.
// C/D layout: VGPR i holds row ((lane>>4)*8 + i), col = colbase + (lane&15).
__device__ __forceinline__ void scatter_bias(float* ldsOut, const v8f& c0, const v8f& c1,
                                             const float* __restrict__ bias, int colbase, int lane) {
    const int col = colbase + (lane & 15);
    const int rb  = (lane >> 4) * 8;
    #pragma unroll
    for (int i = 0; i < 8; ++i) {
        ldsOut[(rb + i) * C_DIM + col]      = c0[i] + bias[col];
        ldsOut[(rb + i) * C_DIM + col + 16] = c1[i] + bias[col + 16];
    }
}

// Per-row mean / rsqrt(var+eps) over the 16x256 LDS tile. tid in [0,256).
__device__ __forceinline__ void row_ln_stats(const float* ldsOut, float* redS, float* redQ,
                                             float* muv, float* rsv, int tid) {
    const int r  = tid >> 4;
    const int cb = (tid & 15) * 16;
    float s = 0.f, q = 0.f;
    #pragma unroll
    for (int j = 0; j < 16; ++j) {
        float x = ldsOut[r * C_DIM + cb + j];
        s += x; q += x * x;
    }
    redS[tid] = s; redQ[tid] = q;
    __syncthreads();
    if ((tid & 15) == 0) {
        float ts = 0.f, tq = 0.f;
        #pragma unroll
        for (int j = 0; j < 16; ++j) { ts += redS[r * 16 + j]; tq += redQ[r * 16 + j]; }
        float mu  = ts * (1.f / 256.f);
        float var = tq * (1.f / 256.f) - mu * mu;
        muv[r] = mu;
        rsv[r] = rsqrtf(var + EPSV);
    }
    __syncthreads();
}

// ---------------------------------------------------------------------------
// Row-block GEMM: out[M x 256] = epilogue( A[M x 256](f32) @ W[256 x 256]^T + bias )
// One workgroup = 16 rows x all 256 cols; 8 waves x 2 N-tiles; K-loop 8 x 32.
// ---------------------------------------------------------------------------
template <int MODE, typename OutT>
__global__ __launch_bounds__(256)
void gemm_rowblock(const float* __restrict__ A, const bf16_t* __restrict__ W,
                   const float* __restrict__ bias,
                   const float* __restrict__ lng, const float* __restrict__ lnb,
                   const float* __restrict__ aux, int auxdiv,
                   OutT* __restrict__ out) {
    __shared__ __align__(32) bf16_t ldsA[512];      // one 16x32 A tile, fragment order
    __shared__ float ldsOut[16 * C_DIM];
    __shared__ float redS[256], redQ[256];
    __shared__ float muv[16], rsv[16];

    const int tid     = threadIdx.x;
    const int lane    = tid & 31;
    const int wave    = tid >> 5;
    const int row0    = blockIdx.x * 16;
    const int colbase = wave * 32;

    v8f c0 = {}; v8f c1 = {};
    const int m_of = lane & 15;
    const int h_of = lane >> 4;

    for (int kk = 0; kk < C_DIM; kk += 32) {
        // Stage A tile (f32 global -> bf16 LDS) in WMMA A-fragment order.
        #pragma unroll
        for (int s = tid; s < 512; s += 256) {
            int l = s >> 4, j = s & 15;
            int m = l & 15, h = l >> 4;
            int k = h * 16 + j;
            ldsA[s] = (bf16_t)A[(size_t)(row0 + m) * C_DIM + kk + k];
        }
        __syncthreads();
        v16bf a = *(const v16bf*)(ldsA + lane * 16);
        const bf16_t* bp = W + (size_t)(colbase + m_of) * C_DIM + kk + h_of * 16;
        v16bf b0 = *(const v16bf*)bp;
        v16bf b1 = *(const v16bf*)(bp + 16 * C_DIM);
        c0 = __builtin_amdgcn_wmma_f32_16x16x32_bf16(false, a, false, b0, (short)0, c0, false, false);
        c1 = __builtin_amdgcn_wmma_f32_16x16x32_bf16(false, a, false, b1, (short)0, c1, false, false);
        __syncthreads();
    }

    scatter_bias(ldsOut, c0, c1, bias, colbase, lane);
    __syncthreads();

    if (MODE == MODE_LN) row_ln_stats(ldsOut, redS, redQ, muv, rsv, tid);

    const int r  = tid >> 4;
    const int cb = (tid & 15) * 16;
    const size_t orow = (size_t)(row0 + r) * C_DIM;
    size_t auxrow = 0;
    if (MODE == MODE_GATE) auxrow = (size_t)((row0 + r) / auxdiv) * C_DIM;
    #pragma unroll
    for (int j = 0; j < 16; ++j) {
        int   c = cb + j;
        float x = ldsOut[r * C_DIM + c];
        float y;
        if (MODE == MODE_LN)        y = (x - muv[r]) * rsv[r] * lng[c] + lnb[c];
        else if (MODE == MODE_GATE) y = x * aux[auxrow + c];
        else                        y = x;
        out[orow + c] = (OutT)y;
    }
}

// ---------------------------------------------------------------------------
// Fused tail: per 16-row block of M = N*K:
//   ig = sigmoid(LN(gate@igW^T+b))          (kept in LDS)
//   ug = sigmoid(LN(gate@ugW^T+b))
//   feat = ug*pout[m/9] + ig*iout           (-> LDS, fragment order, bf16)
//   out = relu(LN(feat@fcW^T + b))
// ---------------------------------------------------------------------------
__global__ __launch_bounds__(256)
void tail_kernel(const bf16_t* __restrict__ gate,
                 const bf16_t* __restrict__ Wig, const bf16_t* __restrict__ Wug,
                 const bf16_t* __restrict__ Wfc,
                 const float* __restrict__ ig_bias, const float* __restrict__ ug_bias,
                 const float* __restrict__ fc_bias,
                 const float* __restrict__ ig_g, const float* __restrict__ ig_b,
                 const float* __restrict__ ug_g, const float* __restrict__ ug_b,
                 const float* __restrict__ fc_g, const float* __restrict__ fc_b,
                 const float* __restrict__ pout, const bf16_t* __restrict__ iout,
                 float* __restrict__ out) {
    __shared__ __align__(32) bf16_t ldsA[8 * 512];   // 16x256 A rows, fragment order
    __shared__ float ldsOut[16 * C_DIM];
    __shared__ float ldsIG[16 * C_DIM];
    __shared__ float redS[256], redQ[256];
    __shared__ float muv[16], rsv[16];

    const int tid     = threadIdx.x;
    const int lane    = tid & 31;
    const int wave    = tid >> 5;
    const int row0    = blockIdx.x * 16;
    const int colbase = wave * 32;
    const int r  = tid >> 4;
    const int cb = (tid & 15) * 16;

    // Stage all gate rows (bf16 ws -> LDS, fragment order).
    #pragma unroll
    for (int s = tid; s < 4096; s += 256) {
        int ks = s >> 9, idx = s & 511;
        int l = idx >> 4, j = idx & 15;
        int m = l & 15, h = l >> 4;
        int kin = h * 16 + j;
        ldsA[s] = gate[(size_t)(row0 + m) * C_DIM + ks * 32 + kin];
    }
    __syncthreads();

    // ---- GEMM 1: input_gate ----
    v8f c0 = {}, c1 = {};
    wave_gemm_256(ldsA, Wig, colbase, lane, c0, c1);
    scatter_bias(ldsOut, c0, c1, ig_bias, colbase, lane);
    __syncthreads();
    row_ln_stats(ldsOut, redS, redQ, muv, rsv, tid);
    #pragma unroll
    for (int j = 0; j < 16; ++j) {
        int   c = cb + j;
        float y = (ldsOut[r * C_DIM + c] - muv[r]) * rsv[r] * ig_g[c] + ig_b[c];
        ldsIG[r * C_DIM + c] = 1.0f / (1.0f + __expf(-y));
    }
    __syncthreads();

    // ---- GEMM 2: update_gate + combine -> feat (fragment-ordered bf16 in ldsA) ----
    c0 = {}; c1 = {};
    wave_gemm_256(ldsA, Wug, colbase, lane, c0, c1);
    scatter_bias(ldsOut, c0, c1, ug_bias, colbase, lane);
    __syncthreads();
    row_ln_stats(ldsOut, redS, redQ, muv, rsv, tid);
    {
        const size_t prow = (size_t)((row0 + r) / 9) * C_DIM;
        const size_t irow = (size_t)(row0 + r) * C_DIM;
        #pragma unroll
        for (int j = 0; j < 16; ++j) {
            int   c  = cb + j;
            float y  = (ldsOut[r * C_DIM + c] - muv[r]) * rsv[r] * ug_g[c] + ug_b[c];
            float ug = 1.0f / (1.0f + __expf(-y));
            float feat = ug * pout[prow + c] + ldsIG[r * C_DIM + c] * (float)iout[irow + c];
            // inverse fragment mapping: (r, c) -> ldsA slot
            int ks  = c >> 5, kin = c & 31;
            int h   = kin >> 4;
            int jj  = kin & 15;
            ldsA[ks * 512 + (h * 16 + r) * 16 + jj] = (bf16_t)feat;
        }
    }
    __syncthreads();

    // ---- GEMM 3: fc + LN + relu -> out ----
    c0 = {}; c1 = {};
    wave_gemm_256(ldsA, Wfc, colbase, lane, c0, c1);
    scatter_bias(ldsOut, c0, c1, fc_bias, colbase, lane);
    __syncthreads();
    row_ln_stats(ldsOut, redS, redQ, muv, rsv, tid);
    {
        const size_t orow = (size_t)(row0 + r) * C_DIM;
        #pragma unroll
        for (int j = 0; j < 16; ++j) {
            int   c = cb + j;
            float y = (ldsOut[r * C_DIM + c] - muv[r]) * rsv[r] * fc_g[c] + fc_b[c];
            out[orow + c] = fmaxf(y, 0.0f);
        }
    }
}

// ---------------------------------------------------------------------------
// Host launcher
// ---------------------------------------------------------------------------
extern "C" void kernel_launch(void* const* d_in, const int* in_sizes, int n_in,
                              void* d_out, int out_size, void* d_ws, size_t ws_size,
                              hipStream_t stream) {
    (void)in_sizes; (void)n_in; (void)out_size; (void)ws_size;
    const float* uf    = (const float*)d_in[0];   // (16384, 256)
    const float* inf   = (const float*)d_in[1];   // (16384, 9, 256)
    const float* dynW  = (const float*)d_in[2];   // (512, 256)
    const float* dynb  = (const float*)d_in[3];
    const float* inpW  = (const float*)d_in[4];   // (512, 256)
    const float* inpb  = (const float*)d_in[5];
    const float* igW   = (const float*)d_in[6];
    const float* igb   = (const float*)d_in[7];
    const float* ugW   = (const float*)d_in[8];
    const float* ugb   = (const float*)d_in[9];
    const float* fcW   = (const float*)d_in[10];
    const float* fcb   = (const float*)d_in[11];
    const float* norm_in_g   = (const float*)d_in[12];
    const float* norm_in_b   = (const float*)d_in[13];
    const float* norm_out_g  = (const float*)d_in[14];
    const float* norm_out_b  = (const float*)d_in[15];
    const float* inorm_in_g  = (const float*)d_in[16];
    const float* inorm_in_b  = (const float*)d_in[17];
    const float* inorm_out_g = (const float*)d_in[18];
    const float* inorm_out_b = (const float*)d_in[19];
    const float* fc_norm_g   = (const float*)d_in[20];
    const float* fc_norm_b   = (const float*)d_in[21];
    float* outp = (float*)d_out;

    const int N  = 16384;
    const int M2 = 16384 * 9;          // 147456

    // Workspace layout
    char* ws = (char*)d_ws;
    bf16_t* Wdyn = (bf16_t*)(ws);                       // 512*256 bf16
    bf16_t* Winp = (bf16_t*)(ws + (size_t)262144);
    bf16_t* Wig  = (bf16_t*)(ws + (size_t)524288);
    bf16_t* Wug  = (bf16_t*)(ws + (size_t)655360);
    bf16_t* Wfc  = (bf16_t*)(ws + (size_t)786432);
    float*  pin  = (float*) (ws + (size_t)1048576);                       // 16 MiB
    float*  pout = (float*) (ws + (size_t)1048576 + 16777216);            // 16 MiB
    bf16_t* gate = (bf16_t*)(ws + (size_t)1048576 + 2 * 16777216);        // 72 MiB
    bf16_t* iout = (bf16_t*)(ws + (size_t)1048576 + 2 * 16777216 + 75497472);

    // Weight conversion (tiny; stays L2-resident thereafter)
    cvt_f32_bf16<<<(131072 + 255) / 256, 256, 0, stream>>>(dynW, Wdyn, 131072);
    cvt_f32_bf16<<<(131072 + 255) / 256, 256, 0, stream>>>(inpW, Winp, 131072);
    cvt_f32_bf16<<<(65536 + 255) / 256, 256, 0, stream>>>(igW, Wig, 65536);
    cvt_f32_bf16<<<(65536 + 255) / 256, 256, 0, stream>>>(ugW, Wug, 65536);
    cvt_f32_bf16<<<(65536 + 255) / 256, 256, 0, stream>>>(fcW, Wfc, 65536);

    // param_in = uf @ dynW[0:256]^T + b          (plain f32)
    gemm_rowblock<MODE_PLAIN, float><<<N / 16, 256, 0, stream>>>(
        uf, Wdyn, dynb, nullptr, nullptr, nullptr, 1, pin);
    // param_out = LN(uf @ dynW[256:512]^T + b)   (norm_out)
    gemm_rowblock<MODE_LN, float><<<N / 16, 256, 0, stream>>>(
        uf, Wdyn + 65536, dynb + 256, norm_out_g, norm_out_b, nullptr, 1, pout);
    // gate = (inf @ inpW[0:256]^T + b) * param_in[m/9]  -> bf16
    gemm_rowblock<MODE_GATE, bf16_t><<<M2 / 16, 256, 0, stream>>>(
        inf, Winp, inpb, nullptr, nullptr, pin, 9, gate);
    // input_out = LN(inf @ inpW[256:512]^T + b)  (inorm_out) -> bf16
    gemm_rowblock<MODE_LN, bf16_t><<<M2 / 16, 256, 0, stream>>>(
        inf, Winp + 65536, inpb + 256, inorm_out_g, inorm_out_b, nullptr, 1, iout);

    // Fused: ig / ug GEMMs + LN + sigmoid + combine + fc GEMM + LN + relu
    tail_kernel<<<M2 / 16, 256, 0, stream>>>(
        gate, Wig, Wug, Wfc,
        igb, ugb, fcb,
        inorm_in_g, inorm_in_b,   // input_gate LN
        norm_in_g, norm_in_b,     // update_gate LN
        fc_norm_g, fc_norm_b,     // final LN
        pout, iout, outp);
}
